// MyGPT_15187004359218
// MI455X (gfx1250) — matrix operand order repaired and checked by reading, not compiled, not run on hardware
//
#include <hip/hip_runtime.h>
#include <hip/hip_bf16.h>
#include <math.h>
#include <stdint.h>

typedef __attribute__((ext_vector_type(16))) _Float16 v16h;
typedef __attribute__((ext_vector_type(8)))  _Float16 f16x8;
typedef __attribute__((ext_vector_type(8)))  float    v8f;

// Model constants
#define GPT_B   2
#define GPT_S   2048
#define GPT_D   768
#define GPT_V   50257
#define GPT_HID 2048
#define GPT_H   12
#define GPT_DH  64
#define GPT_M   (GPT_B * GPT_S)   // 4096 token rows

// LDS leading dim for f16 tiles: 40 f16 = 80B rows -> 16B aligned, and the
// 20-dword row stride hits 16 distinct banks across lanes (gcd(20,64)=4).
#define LDH 40

// ---------------------------------------------------------------------------
// CDNA5 async global->LDS copy (ASYNCcnt-tracked) + wait
// ---------------------------------------------------------------------------
__device__ inline uint32_t lds_off_of(const void* p) {
  return (uint32_t)(uintptr_t)p;   // low 32 bits of flat shared addr = LDS off
}
__device__ inline void async_b128(uint32_t lds_off, const void* gaddr) {
  asm volatile("global_load_async_to_lds_b128 %0, %1, off"
               :: "v"(lds_off), "v"(gaddr) : "memory");
}
__device__ inline void wait_async0() {
  asm volatile("s_wait_asynccnt 0x0" ::: "memory");
}

// ---------------------------------------------------------------------------
// f16 fragment builders (ISA 7.12.2 layouts), all 16B vector loads:
//  A 16x32: lane row; f[0..7] = k 8h..8h+7, f[8..15] = k 16+8h..23+8h
//  B 32x16 (from transposed storage): f[0..7] = k 8h..8h+7, f[8..15] = k 16+8h..23+8h
// ---------------------------------------------------------------------------
__device__ inline v16h frag_a_h(const _Float16* rowp, int half) {
  const f16x8 lo = *(const f16x8*)(rowp + 8 * half);
  const f16x8 hi = *(const f16x8*)(rowp + 8 * half + 16);
  v16h f;
#pragma unroll
  for (int i = 0; i < 8; ++i) { f[i] = lo[i]; f[i + 8] = hi[i]; }
  return f;
}
__device__ inline v16h frag_b_h(const _Float16* rowp, int half) {
  const f16x8 lo = *(const f16x8*)(rowp + 8 * half);
  const f16x8 hi = *(const f16x8*)(rowp + 8 * half + 16);
  v16h f;
#pragma unroll
  for (int i = 0; i < 8; ++i) { f[i] = lo[i]; f[i + 8] = hi[i]; }
  return f;
}

// ---------------------------------------------------------------------------
// One-time weight cast + transpose: Wt[n*K + k] = (f16) W[k*N + n]
// ---------------------------------------------------------------------------
__global__ __launch_bounds__(256) void cast_wT_kernel(
    const float* __restrict__ W, _Float16* __restrict__ Wt, int K, int N) {
  const size_t id = (size_t)blockIdx.x * 256 + threadIdx.x;
  if (id < (size_t)K * N) {
    const int n = (int)(id / (size_t)K);
    const int k = (int)(id % (size_t)K);
    Wt[id] = (_Float16)W[(size_t)k * N + n];
  }
}

// Transpose V per (b,h): vt[(bh*DH + d)*S + s] = vh[(b*S+s)*D + h*DH + d]
__global__ __launch_bounds__(256) void transpose_v_kernel(
    const _Float16* __restrict__ vh, _Float16* __restrict__ vt) {
  const size_t id = (size_t)blockIdx.x * 256 + threadIdx.x;   // over M*D
  if (id >= (size_t)GPT_M * GPT_D) return;
  const int bh = (int)(id / ((size_t)GPT_DH * GPT_S));
  const int rem = (int)(id % ((size_t)GPT_DH * GPT_S));
  const int d = rem / GPT_S;
  const int s = rem % GPT_S;
  const int bb = bh / GPT_H, hh = bh % GPT_H;
  vt[id] = vh[((size_t)bb * GPT_S + s) * GPT_D + hh * GPT_DH + d];
}

// ---------------------------------------------------------------------------
// Embedding + positional encoding -> f16 activations
// ---------------------------------------------------------------------------
__global__ __launch_bounds__(256) void embed_kernel(
    const int* __restrict__ inp, const float* __restrict__ wte,
    _Float16* __restrict__ xh) {
  const int bs  = blockIdx.x;
  const int s   = bs & (GPT_S - 1);
  const int tok = inp[bs];
  for (int d = threadIdx.x; d < GPT_D; d += 256) {
    const float p2   = (float)((d >> 1) << 1);
    const float freq = expf(p2 * (-9.210340371976184f / (float)GPT_D));
    const float ang  = (float)s * freq;
    const float pe   = (d & 1) ? cosf(ang) : sinf(ang);
    xh[(size_t)bs * GPT_D + d] = (_Float16)(wte[(size_t)tok * GPT_D + d] + pe);
  }
}

// ---------------------------------------------------------------------------
// GEMM: out = act(A_h[M,K] @ Wt_h[N,K]^T + bias) * oscale
// Block 128 thr (2x2 waves) -> 64x64 tile; 32x32/wave; 4 WMMA per k-step.
// Double-buffered async b128 staging; f16 everywhere in the hot loop.
// ---------------------------------------------------------------------------
__global__ __launch_bounds__(128) void gemm_wmma_kernel(
    const _Float16* __restrict__ A, const _Float16* __restrict__ Wt,
    const float* __restrict__ bias, float* __restrict__ Cf,
    _Float16* __restrict__ Ch, int M, int K, int N, int relu, float oscale) {
  __shared__ _Float16 Ash[2][64 * LDH];   // [row][k]
  __shared__ _Float16 Bsh[2][64 * LDH];   // [col][k]

  const int tid    = threadIdx.x;
  const int wid    = tid >> 5;
  const int lane   = tid & 31;
  const int half   = lane >> 4;
  const int ln     = lane & 15;
  const int wave_m = wid >> 1;
  const int wave_n = wid & 1;
  const int m0     = blockIdx.y * 64;
  const int n0     = blockIdx.x * 64;

  const uint32_t aoff[2] = { lds_off_of(&Ash[0][0]), lds_off_of(&Ash[1][0]) };
  const uint32_t boff[2] = { lds_off_of(&Bsh[0][0]), lds_off_of(&Bsh[1][0]) };

  auto stage = [&](int buf, int k0) {
    // A: 64 rows x 32 k f16 = 256 x 16B chunks, 2 per thread
#pragma unroll
    for (int i = 0; i < 2; ++i) {
      const int e  = tid + 128 * i;        // 0..255
      const int r  = e >> 2;               // row 0..63
      const int c8 = (e & 3) << 3;         // k element 0,8,16,24
      async_b128(aoff[buf] + (uint32_t)((r * LDH + c8) << 1),
                 &A[(size_t)(m0 + r) * K + k0 + c8]);
    }
    // B: 64 cols x 32 k f16 from transposed weights (contiguous k)
#pragma unroll
    for (int i = 0; i < 2; ++i) {
      const int e  = tid + 128 * i;
      const int c  = e >> 2;               // col 0..63
      const int k8 = (e & 3) << 3;
      const int gc0 = n0 + c;
      const int gc  = (gc0 < N) ? gc0 : (N - 1);   // branch-free edge clamp
      async_b128(boff[buf] + (uint32_t)((c * LDH + k8) << 1),
                 &Wt[(size_t)gc * K + k0 + k8]);
    }
  };

  v8f acc[2][2];
#pragma unroll
  for (int mt = 0; mt < 2; ++mt)
#pragma unroll
    for (int nt = 0; nt < 2; ++nt) acc[mt][nt] = (v8f){};

  stage(0, 0);
  int buf = 0;
  for (int k0 = 0; k0 < K; k0 += 32, buf ^= 1) {
    wait_async0();
    __syncthreads();
    if (k0 + 32 < K) stage(buf ^ 1, k0 + 32);

    v16h af[2], bf[2];
#pragma unroll
    for (int mt = 0; mt < 2; ++mt)
      af[mt] = frag_a_h(&Ash[buf][(wave_m * 32 + mt * 16 + ln) * LDH], half);
#pragma unroll
    for (int nt = 0; nt < 2; ++nt)
      bf[nt] = frag_b_h(&Bsh[buf][(wave_n * 32 + nt * 16 + ln) * LDH], half);

#pragma unroll
    for (int mt = 0; mt < 2; ++mt)
#pragma unroll
      for (int nt = 0; nt < 2; ++nt)
        acc[mt][nt] = __builtin_amdgcn_wmma_f32_16x16x32_f16(
            false, af[mt], false, bf[nt], (short)0, acc[mt][nt], false, false);
  }

  // epilogue: C layout VGPR j -> row j+8*half, col = lane&15
#pragma unroll
  for (int mt = 0; mt < 2; ++mt) {
#pragma unroll
    for (int nt = 0; nt < 2; ++nt) {
      const int col = n0 + wave_n * 32 + nt * 16 + ln;
      if (col < N) {
        const float bv = bias[col];
#pragma unroll
        for (int j = 0; j < 8; ++j) {
          const int row = m0 + wave_m * 32 + mt * 16 + j + 8 * half;
          float vv = acc[mt][nt][j] + bv;
          if (relu) vv = fmaxf(vv, 0.f);
          if (Cf) Cf[(size_t)row * N + col] = vv;
          if (Ch) Ch[(size_t)row * N + col] = (_Float16)(vv * oscale);
        }
      }
    }
  }
}

// ---------------------------------------------------------------------------
// Flash-style causal attention, all-f16 operands.
// One wave = one 16-query tile of one (b,h); 32-key blocks.
// ---------------------------------------------------------------------------
__global__ __launch_bounds__(128) void flash_attn_kernel(
    const _Float16* __restrict__ q, const _Float16* __restrict__ k,
    const _Float16* __restrict__ vt, _Float16* __restrict__ ctx) {
  __shared__ _Float16 Pl[4][16 * LDH];

  const int tid  = threadIdx.x;
  const int wid  = tid >> 5;
  const int lane = tid & 31;
  const int half = lane >> 4;
  const int ln   = lane & 15;

  const int wg = blockIdx.x * 4 + wid;
  const int bh = wg >> 7;
  const int qt = wg & 127;
  const int bb = bh / GPT_H;
  const int hh = bh % GPT_H;
  const int q0 = qt * 16;

  const size_t base   = ((size_t)bb * GPT_S) * GPT_D + (size_t)hh * GPT_DH;
  const size_t vtbase = ((size_t)bh * GPT_DH) * GPT_S;

  // Q fragments (q was pre-scaled by 1/8 in its GEMM epilogue)
  v16h qa[2];
#pragma unroll
  for (int ks = 0; ks < 2; ++ks)
    qa[ks] = frag_a_h(&q[base + (size_t)(q0 + ln) * GPT_D + ks * 32], half);

  v8f acc[4];
#pragma unroll
  for (int t = 0; t < 4; ++t) acc[t] = (v8f){};
  float mr[8], lr[8];
#pragma unroll
  for (int j = 0; j < 8; ++j) { mr[j] = -1e30f; lr[j] = 0.f; }

  _Float16* myP = &Pl[wid][0];

  for (int kb = 0; kb <= q0 + 15; kb += 32) {
    v8f s[2]; s[0] = (v8f){}; s[1] = (v8f){};
#pragma unroll
    for (int ns = 0; ns < 2; ++ns) {
#pragma unroll
      for (int ks = 0; ks < 2; ++ks) {
        const v16h bf = frag_b_h(
            &k[base + (size_t)(kb + ns * 16 + ln) * GPT_D + ks * 32], half);
        s[ns] = __builtin_amdgcn_wmma_f32_16x16x32_f16(false, qa[ks], false, bf,
                                                       (short)0, s[ns], false, false);
      }
    }

    float scl[8];
#pragma unroll
    for (int j = 0; j < 8; ++j) {
      const int row = q0 + j + 8 * half;
#pragma unroll
      for (int ns = 0; ns < 2; ++ns) {
        const int col = kb + ns * 16 + ln;
        if (col > row) s[ns][j] = -1e30f;
      }
      float mx = fmaxf(s[0][j], s[1][j]);
      mx = fmaxf(mx, __shfl_xor(mx, 1));
      mx = fmaxf(mx, __shfl_xor(mx, 2));
      mx = fmaxf(mx, __shfl_xor(mx, 4));
      mx = fmaxf(mx, __shfl_xor(mx, 8));
      const float mnew = fmaxf(mr[j], mx);

      const float p0 = expf(s[0][j] - mnew);
      const float p1 = expf(s[1][j] - mnew);
      s[0][j] = p0; s[1][j] = p1;
      float sm = p0 + p1;
      sm += __shfl_xor(sm, 1);
      sm += __shfl_xor(sm, 2);
      sm += __shfl_xor(sm, 4);
      sm += __shfl_xor(sm, 8);

      scl[j] = expf(mr[j] - mnew);
      lr[j]  = lr[j] * scl[j] + sm;
      mr[j]  = mnew;
    }
#pragma unroll
    for (int t = 0; t < 4; ++t)
#pragma unroll
      for (int j = 0; j < 8; ++j) acc[t][j] *= scl[j];

    // transpose P (C-layout -> A-layout) via per-wave LDS scratch
#pragma unroll
    for (int ns = 0; ns < 2; ++ns)
#pragma unroll
      for (int j = 0; j < 8; ++j)
        myP[(j + 8 * half) * LDH + ns * 16 + ln] = (_Float16)s[ns][j];
    __builtin_amdgcn_wave_barrier();

    const v16h pa = frag_a_h(&myP[ln * LDH], half);

#pragma unroll
    for (int t = 0; t < 4; ++t) {
      const v16h vf = frag_b_h(
          &vt[vtbase + (size_t)(t * 16 + ln) * GPT_S + kb], half);
      acc[t] = __builtin_amdgcn_wmma_f32_16x16x32_f16(false, pa, false, vf,
                                                      (short)0, acc[t], false, false);
    }
  }

#pragma unroll
  for (int j = 0; j < 8; ++j) {
    const float inv = 1.0f / lr[j];
    const int row = q0 + j + 8 * half;
#pragma unroll
    for (int t = 0; t < 4; ++t)
      ctx[base + (size_t)row * GPT_D + t * 16 + ln] = (_Float16)(acc[t][j] * inv);
  }
}

// ---------------------------------------------------------------------------
// LayerNorm over D=768 (f32 in, f16 out)
// ---------------------------------------------------------------------------
__global__ __launch_bounds__(256) void layernorm_kernel(
    const float* __restrict__ h, const float* __restrict__ g,
    const float* __restrict__ bta, _Float16* __restrict__ o) {
  __shared__ float red[256];
  const int row = blockIdx.x;
  const float* hr = h + (size_t)row * GPT_D;

  float s = 0.f;
  for (int d = threadIdx.x; d < GPT_D; d += 256) s += hr[d];
  red[threadIdx.x] = s; __syncthreads();
  for (int off = 128; off > 0; off >>= 1) {
    if (threadIdx.x < off) red[threadIdx.x] += red[threadIdx.x + off];
    __syncthreads();
  }
  const float mu = red[0] / (float)GPT_D;
  __syncthreads();

  float vs = 0.f;
  for (int d = threadIdx.x; d < GPT_D; d += 256) {
    const float t = hr[d] - mu; vs += t * t;
  }
  red[threadIdx.x] = vs; __syncthreads();
  for (int off = 128; off > 0; off >>= 1) {
    if (threadIdx.x < off) red[threadIdx.x] += red[threadIdx.x + off];
    __syncthreads();
  }
  const float rstd = rsqrtf(red[0] / (float)GPT_D + 1e-5f);
  for (int d = threadIdx.x; d < GPT_D; d += 256)
    o[(size_t)row * GPT_D + d] = (_Float16)((hr[d] - mu) * rstd * g[d] + bta[d]);
}

// ---------------------------------------------------------------------------
// Loss: per-row NLL then mean
// ---------------------------------------------------------------------------
__global__ __launch_bounds__(256) void loss_row_kernel(
    const float* __restrict__ logits, const int* __restrict__ targets,
    float* __restrict__ row_nll) {
  __shared__ float red[256];
  const int row = blockIdx.x;
  const float* lrow = logits + (size_t)row * GPT_V;

  float mx = -1e30f;
  for (int j = threadIdx.x; j < GPT_V; j += 256) mx = fmaxf(mx, lrow[j]);
  red[threadIdx.x] = mx; __syncthreads();
  for (int off = 128; off > 0; off >>= 1) {
    if (threadIdx.x < off)
      red[threadIdx.x] = fmaxf(red[threadIdx.x], red[threadIdx.x + off]);
    __syncthreads();
  }
  mx = red[0]; __syncthreads();

  float sm = 0.f;
  for (int j = threadIdx.x; j < GPT_V; j += 256) sm += expf(lrow[j] - mx);
  red[threadIdx.x] = sm; __syncthreads();
  for (int off = 128; off > 0; off >>= 1) {
    if (threadIdx.x < off) red[threadIdx.x] += red[threadIdx.x + off];
    __syncthreads();
  }
  if (threadIdx.x == 0)
    row_nll[row] = mx + logf(red[0]) - lrow[targets[row]];
}

__global__ __launch_bounds__(256) void loss_mean_kernel(
    const float* __restrict__ row_nll, float* __restrict__ out_loss) {
  __shared__ float red[256];
  float s = 0.f;
  for (int i = threadIdx.x; i < GPT_M; i += 256) s += row_nll[i];
  red[threadIdx.x] = s; __syncthreads();
  for (int off = 128; off > 0; off >>= 1) {
    if (threadIdx.x < off) red[threadIdx.x] += red[threadIdx.x + off];
    __syncthreads();
  }
  if (threadIdx.x == 0) *out_loss = red[0] / (float)GPT_M;
}

// ---------------------------------------------------------------------------
// Launcher
// ---------------------------------------------------------------------------
extern "C" void kernel_launch(void* const* d_in, const int* in_sizes, int n_in,
                              void* d_out, int out_size, void* d_ws, size_t ws_size,
                              hipStream_t stream) {
  (void)in_sizes; (void)n_in; (void)out_size; (void)ws_size;

  const int*   inputs  = (const int*)  d_in[0];
  const int*   targets = (const int*)  d_in[1];
  const float* wte     = (const float*)d_in[2];
  const float* wq      = (const float*)d_in[3];
  const float* bq      = (const float*)d_in[4];
  const float* wk      = (const float*)d_in[5];
  const float* bk      = (const float*)d_in[6];
  const float* wv      = (const float*)d_in[7];
  const float* bv      = (const float*)d_in[8];
  const float* wo      = (const float*)d_in[9];
  const float* bo      = (const float*)d_in[10];
  const float* w1      = (const float*)d_in[11];
  const float* b1      = (const float*)d_in[12];
  const float* w2      = (const float*)d_in[13];
  const float* b2      = (const float*)d_in[14];
  const float* ln_g    = (const float*)d_in[15];
  const float* ln_b    = (const float*)d_in[16];
  const float* w_out   = (const float*)d_in[17];
  const float* b_out   = (const float*)d_in[18];

  float* logits = (float*)d_out;
  float* loss   = logits + (size_t)GPT_M * GPT_V;

  const size_t MD  = (size_t)GPT_M * GPT_D;
  const size_t MH  = (size_t)GPT_M * GPT_HID;
  const size_t DD  = (size_t)GPT_D * GPT_D;
  const size_t DHd = (size_t)GPT_D * GPT_HID;
  const size_t DV  = (size_t)GPT_D * GPT_V;

  // f32 region
  float* fw      = (float*)d_ws;
  float* hb      = fw;              // [M,D] pre-LN activations (f32)
  float* row_nll = hb + MD;         // [M]
  // f16 region (starts 16B aligned)
  _Float16* hw   = (_Float16*)(row_nll + GPT_M);
  _Float16* xh   = hw;              // [M,D]
  _Float16* qh   = xh  + MD;
  _Float16* kh   = qh  + MD;
  _Float16* vh   = kh  + MD;
  _Float16* vth  = vh  + MD;        // [B*H, DH, S] transposed V
  _Float16* ctxh = vth + MD;
  _Float16* aoh  = ctxh+ MD;
  _Float16* hnh  = aoh + MD;
  _Float16* h1h  = hnh + MD;        // [M,HID]
  _Float16* wqt  = h1h + MH;        // transposed f16 weights [N,K]
  _Float16* wkt  = wqt + DD;
  _Float16* wvt  = wkt + DD;
  _Float16* wot  = wvt + DD;
  _Float16* w1t  = wot + DD;        // [HID, D]
  _Float16* w2t  = w1t + DHd;       // [D, HID]
  _Float16* woutt= w2t + DHd;       // [V, D]

  const dim3 blk128(128), blk256(256);
  const dim3 gD(GPT_D / 64, GPT_M / 64);
  const dim3 gHID(GPT_HID / 64, GPT_M / 64);
  const dim3 gV((GPT_V + 63) / 64, GPT_M / 64);

  // 0. one-time weight cast+transpose (f32 [K,N] -> f16 [N,K])
  cast_wT_kernel<<<(int)((DD  + 255) / 256), blk256, 0, stream>>>(wq,    wqt,  GPT_D,   GPT_D);
  cast_wT_kernel<<<(int)((DD  + 255) / 256), blk256, 0, stream>>>(wk,    wkt,  GPT_D,   GPT_D);
  cast_wT_kernel<<<(int)((DD  + 255) / 256), blk256, 0, stream>>>(wv,    wvt,  GPT_D,   GPT_D);
  cast_wT_kernel<<<(int)((DD  + 255) / 256), blk256, 0, stream>>>(wo,    wot,  GPT_D,   GPT_D);
  cast_wT_kernel<<<(int)((DHd + 255) / 256), blk256, 0, stream>>>(w1,    w1t,  GPT_D,   GPT_HID);
  cast_wT_kernel<<<(int)((DHd + 255) / 256), blk256, 0, stream>>>(w2,    w2t,  GPT_HID, GPT_D);
  cast_wT_kernel<<<(int)((DV  + 255) / 256), blk256, 0, stream>>>(w_out, woutt,GPT_D,   GPT_V);

  // 1. embedding + PE
  embed_kernel<<<GPT_M, blk256, 0, stream>>>(inputs, wte, xh);

  // 2. QKV projections (q pre-scaled by 1/sqrt(64))
  gemm_wmma_kernel<<<gD, blk128, 0, stream>>>(xh, wqt, bq, nullptr, qh,
                                              GPT_M, GPT_D, GPT_D, 0, 0.125f);
  gemm_wmma_kernel<<<gD, blk128, 0, stream>>>(xh, wkt, bk, nullptr, kh,
                                              GPT_M, GPT_D, GPT_D, 0, 1.0f);
  gemm_wmma_kernel<<<gD, blk128, 0, stream>>>(xh, wvt, bv, nullptr, vh,
                                              GPT_M, GPT_D, GPT_D, 0, 1.0f);
  transpose_v_kernel<<<(int)((MD + 255) / 256), blk256, 0, stream>>>(vh, vth);

  // 3. causal flash attention
  flash_attn_kernel<<<768, blk128, 0, stream>>>(qh, kh, vth, ctxh);

  // 4. output projection + FFN
  gemm_wmma_kernel<<<gD, blk128, 0, stream>>>(ctxh, wot, bo, nullptr, aoh,
                                              GPT_M, GPT_D, GPT_D, 0, 1.0f);
  gemm_wmma_kernel<<<gHID, blk128, 0, stream>>>(aoh, w1t, b1, nullptr, h1h,
                                                GPT_M, GPT_D, GPT_HID, 1, 1.0f);
  gemm_wmma_kernel<<<gD, blk128, 0, stream>>>(h1h, w2t, b2, hb, nullptr,
                                              GPT_M, GPT_HID, GPT_D, 0, 1.0f);

  // 5. LayerNorm (f32 -> f16)
  layernorm_kernel<<<GPT_M, blk256, 0, stream>>>(hb, ln_g, ln_b, hnh);

  // 6. logits (dominant GEMM; f16 weights fit in 192MB L2)
  gemm_wmma_kernel<<<gV, blk128, 0, stream>>>(hnh, woutt, b_out, logits, nullptr,
                                              GPT_M, GPT_D, GPT_V, 0, 1.0f);

  // 7. loss
  loss_row_kernel<<<GPT_M, blk256, 0, stream>>>(logits, targets, row_nll);
  loss_mean_kernel<<<1, blk256, 0, stream>>>(row_nll, loss);
}
